// KerasRNNCellWrapper_6055903887832
// MI455X (gfx1250) — matrix operand order repaired and checked by reading, not compiled, not run on hardware
//
#include <hip/hip_runtime.h>
#include <hip/hip_bf16.h>

// ---------------------------------------------------------------------------
// Keras SimpleRNN on MI455X (gfx1250, wave32, WMMA bf16 -> f32 accum)
//   B=64, T=512, D=256, H=1024, O=128
// Pipeline:
//   K0  convert inputs -> bf16, weights -> transposed bf16 [N][K]
//   K1  xproj[t][b][:] = X @ Wx + bias            (fp32 in workspace)
//   K2  persistent (32 blocks): 512 steps h = tanh(xproj_t + h @ Wh),
//       Wh slab cached in LDS (XOR-swizzled, conflict-free ds_load_b128),
//       device-wide barrier per step
//   K3  y = hs @ Wo + bo                          (fp32 to d_out, [B][T][O])
// ---------------------------------------------------------------------------

typedef __attribute__((ext_vector_type(16))) __bf16 v16bf;
typedef __attribute__((ext_vector_type(8)))  __bf16 v8bf;
typedef __attribute__((ext_vector_type(8)))  float  v8f;

union FragU { v16bf v; v8bf h[2]; };

// A-fragment (16x32 bf16, M rows x K cols) from row-major [rows][ld] bf16.
// Layout: lane l: M = row0 + l%16 ; element e:
//   K = k0 + 16*(l/16) + e
__device__ __forceinline__ v16bf load_fragA(const __bf16* p, int ld, int row0,
                                            int k0, int lane) {
  const __bf16* base =
      p + (size_t)(row0 + (lane & 15)) * ld + k0 + ((lane >> 4) << 4);
  FragU f;
  f.h[0] = *(const v8bf*)(base);        // elems 0..7
  f.h[1] = *(const v8bf*)(base + 8);    // elems 8..15
  return f.v;
}

// B-fragment (32x16 bf16, K rows x N cols) from *transposed* weights WT[N][ld].
// Layout: lane l: N = n0 + l%16 ; element e: K = k0 + 16*(l/16) + e
__device__ __forceinline__ v16bf load_fragB(const __bf16* wt, int ld, int n0,
                                            int k0, int lane) {
  const __bf16* base =
      wt + (size_t)(n0 + (lane & 15)) * ld + k0 + ((lane >> 4) << 4);
  FragU f;
  f.h[0] = *(const v8bf*)(base);        // elems 0..7
  f.h[1] = *(const v8bf*)(base + 8);    // elems 8..15
  return f.v;
}

// B-fragment from LDS slab [32 rows][1024 bf16], XOR-swizzled in 16B chunks:
// chunk c of row r stored at chunk (c ^ (r & 15)). Rows are 2048B apart, so
// the swizzle puts lane r's b128 on banks 4r..4r+3 -> conflict-free.
__device__ __forceinline__ v16bf load_fragB_lds(const __bf16* s, int nrel0,
                                                int k0, int lane) {
  const int r  = nrel0 + (lane & 15);
  const int rx = r & 15;
  const int kk = k0 + ((lane >> 4) << 4);  // K offset of element 0
  const int c0 = kk >> 3;                  // 16B chunk index (8 bf16)
  const __bf16* row = s + r * 1024;
  FragU f;
  f.h[0] = *(const v8bf*)(row + ((c0 ^ rx) << 3));        // elems 0..7
  f.h[1] = *(const v8bf*)(row + (((c0 + 1) ^ rx) << 3));  // elems 8..15
  return f.v;
}

__device__ __forceinline__ v8f wmma_bf16(v16bf a, v16bf b, v8f c) {
  // (neg_a, A, neg_b, B, c_mod, C, reuse_a, reuse_b)
  return __builtin_amdgcn_wmma_f32_16x16x32_bf16(false, a, false, b, (short)0,
                                                 c, false, false);
}

// ------------------------- K0: converts ------------------------------------
__global__ void cvt_copy(const float* __restrict__ in, __bf16* __restrict__ out,
                         int n) {
  int i = blockIdx.x * blockDim.x + threadIdx.x;
  if (i < n) out[i] = (__bf16)in[i];
}

// in: [K][N] fp32 row-major -> out: [N][K] bf16 row-major
__global__ void cvt_transpose(const float* __restrict__ in,
                              __bf16* __restrict__ out, int K, int N) {
  int i = blockIdx.x * blockDim.x + threadIdx.x;
  if (i < K * N) {
    int k = i / N, n = i - k * N;
    out[(size_t)n * K + k] = (__bf16)in[i];
  }
}

__global__ void init_counter(unsigned* c) {
  if (threadIdx.x == 0) *c = 0u;
}

// ------------------------- K1: xproj = X @ Wx + b --------------------------
// XB: [B*T=32768][D=256] bf16 (row m = b*T + t).  WxT: [H=1024][D=256] bf16.
// xproj: [T][B][H] fp32.
__global__ void xproj_gemm(const __bf16* __restrict__ XB,
                           const __bf16* __restrict__ WxT,
                           const float* __restrict__ bias,
                           float* __restrict__ xproj) {
  const int lane = threadIdx.x & 31, w = threadIdx.x >> 5;
  const int tid = blockIdx.x * 8 + w;     // 131072 wave-tiles
  const int ntile = tid & 63;             // 64 N-tiles (H/16)
  const int mtile = tid >> 6;             // 2048 M-tiles
  const int m0 = mtile * 16, n0 = ntile * 16;
  v8f acc = {};
#pragma unroll
  for (int kk = 0; kk < 8; ++kk) {        // K = 256 = 8 * 32
    v16bf a = load_fragA(XB, 256, m0, kk * 32, lane);
    v16bf b = load_fragB(WxT, 256, n0, kk * 32, lane);
    acc = wmma_bf16(a, b, acc);
  }
  const int n = n0 + (lane & 15);
  const float bv = bias[n];
  const int mbase = m0 + ((lane >> 4) << 3);
#pragma unroll
  for (int r = 0; r < 8; ++r) {
    int m = mbase + r;                    // m = b*T + t
    int bb = m >> 9;                      // / 512
    int tt = m & 511;
    xproj[((size_t)tt * 64 + bb) * 1024 + n] = acc[r] + bv;
  }
}

// ------------------------- K2: persistent recurrence -----------------------
// 32 blocks x 256 threads. Block owns 32 columns of H; its WhT slab
// (32 rows x 1024 K = 64 KB) lives in LDS for all 512 steps.
// Wave w of 8: m-tile = w>>1 (batch), n-tile = w&1 -> one 16x16 accumulator.
// hs: [T][B][H] bf16. h_{t-1} read from hs[t-1]; h_t (post-tanh) to hs[t].
__global__ void __launch_bounds__(256, 1)
rnn_steps(const float* __restrict__ xproj, const __bf16* __restrict__ WhT,
          __bf16* __restrict__ hs, unsigned* counter) {
  __shared__ __bf16 sWh[32 * 1024];       // 64 KB, XOR-swizzled 16B chunks

  const int lane = threadIdx.x & 31, w = threadIdx.x >> 5;
  const int m0 = (w >> 1) * 16;           // batch tile (M)
  const int nrel0 = (w & 1) * 16;         // column tile within block slab
  const int n0g = blockIdx.x * 32 + nrel0;
  const int H = 1024;
  const int nG = n0g + (lane & 15);       // global H column of this lane
  const int mbase = m0 + ((lane >> 4) << 3);

  // ---- one-time LDS fill: 32 contiguous WhT rows, swizzled 16B chunks ----
  {
    const __bf16* slab = WhT + (size_t)blockIdx.x * 32 * 1024;
#pragma unroll
    for (int i = threadIdx.x; i < 4096; i += 256) {  // 4096 x 16B = 64 KB
      int r = i >> 7;                     // row 0..31
      int c = i & 127;                    // 16B chunk 0..127
      v8bf val = *(const v8bf*)(slab + (size_t)r * 1024 + c * 8);
      *(v8bf*)(sWh + r * 1024 + ((c ^ (r & 15)) << 3)) = val;
    }
  }
  __syncthreads();

  for (int t = 0; t < 512; ++t) {
    const float* xp = xproj + (size_t)t * 64 * H;
    v8f acc;
#pragma unroll
    for (int r = 0; r < 8; ++r)           // fp32 xproj straight into accum
      acc[r] = xp[(size_t)(mbase + r) * H + nG];
    // speculative prefetch of next step's xproj (gfx1250 global_prefetch)
    __builtin_prefetch(xp + (size_t)(64 + mbase) * H + nG, 0, 1);

    if (t > 0) {                          // h0 == 0 -> skip GEMM at t==0
      const __bf16* hprev = hs + (size_t)(t - 1) * 64 * H;
#pragma unroll 4
      for (int kk = 0; kk < 32; ++kk) {   // K = 1024 = 32 * 32
        const int k0 = kk * 32;
        v16bf a = load_fragA(hprev, H, m0, k0, lane);   // global (L2)
        v16bf b = load_fragB_lds(sWh, nrel0, k0, lane); // LDS, no conflicts
        acc = wmma_bf16(a, b, acc);
      }
    }
    __bf16* hout = hs + (size_t)t * 64 * H;
#pragma unroll
    for (int r = 0; r < 8; ++r)
      hout[(size_t)(mbase + r) * H + nG] = (__bf16)tanhf(acc[r]);

    // ---- device-wide barrier: monotonic counter, one arrival per block ----
    __threadfence();
    __syncthreads();
    if (threadIdx.x == 0) {
      atomicAdd(counter, 1u);
      const unsigned target = 32u * (unsigned)(t + 1);
      while (atomicAdd(counter, 0u) < target) __builtin_amdgcn_s_sleep(2);
    }
    __syncthreads();
    __threadfence();
  }
}

// ------------------------- K3: y = hs @ Wo + bo ----------------------------
// hs rows m = t*64 + b; out is [B][T][O] fp32. WoT: [O=128][H=1024] bf16.
__global__ void out_gemm(const __bf16* __restrict__ hs,
                         const __bf16* __restrict__ WoT,
                         const float* __restrict__ bo,
                         float* __restrict__ out) {
  const int lane = threadIdx.x & 31, w = threadIdx.x >> 5;
  const int tid = blockIdx.x * 8 + w;     // 16384 wave-tiles
  const int ntile = tid & 7;              // 8 N-tiles (O/16)
  const int mtile = tid >> 3;             // 2048 M-tiles
  const int m0 = mtile * 16, n0 = ntile * 16;
  v8f acc = {};
#pragma unroll 8
  for (int kk = 0; kk < 32; ++kk) {       // K = 1024
    v16bf a = load_fragA(hs, 1024, m0, kk * 32, lane);
    v16bf b = load_fragB(WoT, 1024, n0, kk * 32, lane);
    acc = wmma_bf16(a, b, acc);
  }
  const int o = n0 + (lane & 15);
  const float bv = bo[o];
  const int mbase = m0 + ((lane >> 4) << 3);
#pragma unroll
  for (int r = 0; r < 8; ++r) {
    int m = mbase + r;                    // m = t*64 + b
    int tt = m >> 6, bb = m & 63;
    out[((size_t)bb * 512 + tt) * 128 + o] = acc[r] + bv;
  }
}

// ---------------------------------------------------------------------------
extern "C" void kernel_launch(void* const* d_in, const int* in_sizes, int n_in,
                              void* d_out, int out_size, void* d_ws,
                              size_t ws_size, hipStream_t stream) {
  const float* inputs = (const float*)d_in[0];  // [64][512][256]
  const float* Wx     = (const float*)d_in[1];  // [256][1024]
  const float* Wh     = (const float*)d_in[2];  // [1024][1024]
  const float* bias   = (const float*)d_in[3];  // [1024]
  const float* Wo     = (const float*)d_in[4];  // [1024][128]
  const float* bo     = (const float*)d_in[5];  // [128]
  float* out = (float*)d_out;                   // [64][512][128]

  char* ws = (char*)d_ws;
  // workspace layout (256B aligned offsets), total ~211 MB
  __bf16* XB    = (__bf16*)(ws + 0);            //  16,777,216 B
  __bf16* WxT   = (__bf16*)(ws + 16777216);     //     524,288 B
  __bf16* WhT   = (__bf16*)(ws + 17301504);     //   2,097,152 B
  __bf16* WoT   = (__bf16*)(ws + 19398656);     //     262,144 B
  float*  xproj = (float*)(ws + 19660800);      // 134,217,728 B
  __bf16* hs    = (__bf16*)(ws + 153878528);    //  67,108,864 B
  unsigned* ctr = (unsigned*)(ws + 220987392);  //           4 B

  init_counter<<<1, 64, 0, stream>>>(ctr);

  // K0: conversions / transposes
  cvt_copy<<<32768, 256, 0, stream>>>(inputs, XB, 64 * 512 * 256);
  cvt_transpose<<<1024, 256, 0, stream>>>(Wx, WxT, 256, 1024);
  cvt_transpose<<<4096, 256, 0, stream>>>(Wh, WhT, 1024, 1024);
  cvt_transpose<<<512, 256, 0, stream>>>(Wo, WoT, 1024, 128);

  // K1: input projection (2048 M-tiles x 64 N-tiles, 8 wave-tiles per block)
  xproj_gemm<<<16384, 256, 0, stream>>>(XB, WxT, bias, xproj);

  // K2: persistent recurrence, 32 blocks, Wh slab in LDS, barrier per step
  rnn_steps<<<32, 256, 0, stream>>>(xproj, WhT, hs, ctr);

  // K3: output projection (2048 M-tiles x 8 N-tiles)
  out_gemm<<<2048, 256, 0, stream>>>(hs, WoT, bo, out);
}
